// UNet5_21698174779792
// MI455X (gfx1250) — matrix operand order repaired and checked by reading, not compiled, not run on hardware
//
#include <hip/hip_runtime.h>

typedef __attribute__((ext_vector_type(16))) _Float16 v16h;
typedef __attribute__((ext_vector_type(8)))  float    v8f;

#define CIN   16
#define COUT  32
#define KS    4
#define TAPS  64            // KS^3
#define RDIM  (TAPS * CIN)  // 1024 reduction elements per voxel
#define VPB   16            // voxels per block (= WMMA M)
#define NTHREADS 256

// Dynamic LDS layout:
//   accLds : VPB * RDIM fp32                  = 65536 B
//   wLds   : RDIM * COUT f16 (B-frag swizzle) = 65536 B
//   dLds   : 2 * 16 * 16 fp32 partial D       =  2048 B
//   denLds : VPB fp32                          =    64 B
#define ACC_BYTES (VPB * RDIM * 4)
#define W_BYTES   (RDIM * COUT * 2)
#define D_BYTES   (2 * 16 * 16 * 4)
#define LDS_BYTES (ACC_BYTES + W_BYTES + D_BYTES + VPB * 4)

// ---------------------------------------------------------------------------
// Prep kernel: convert fp32 filter bank [64][16][32] once into a
// 16-bit B-matrix (K=32, N=16) fragment layout with contiguous k order, as f16, into d_ws.
// Layout: lanes 0-15 hold K 0-15, lanes 16-31 hold K 16-31, f16 slot = K%16.
// ---------------------------------------------------------------------------
__global__ void weights_swizzle_kernel(const float* __restrict__ kern,
                                       _Float16* __restrict__ wsw)
{
    for (int idx = blockIdx.x * blockDim.x + threadIdx.x;
         idx < RDIM * COUT; idx += gridDim.x * blockDim.x) {
        const int co   = idx & (COUT - 1);
        const int r    = idx >> 5;          // 0..1023
        const int kk   = r >> 5;            // K-block (32 elems)
        const int rl   = r & 31;            // K within block
        const int half = rl >> 4;
        const int slot = rl & 15;
        const int lane = half * 16 + (co & 15);
        const int ntile = co >> 4;
        wsw[((kk * 2 + ntile) * 32 + lane) * 16 + slot] = (_Float16)kern[idx];
    }
}

__global__ __launch_bounds__(NTHREADS)
void cconv_fused_kernel(const float* __restrict__ feats,
                        const float* __restrict__ inp_points,
                        const float* __restrict__ out_points,
                        const float* __restrict__ out_extents,
                        const float* __restrict__ scale_compat,
                        const float* __restrict__ ndist,
                        const int*   __restrict__ nidx,
                        const int*   __restrict__ rowsplits,
                        const float* __restrict__ kern,
                        const float* __restrict__ bias,
                        const _Float16* __restrict__ wpre,  // pre-swizzled f16 (or null)
                        float* __restrict__ out,
                        int n_out)
{
    extern __shared__ char smem[];
    float*    accLds = (float*)smem;
    _Float16* wLds   = (_Float16*)(smem + ACC_BYTES);
    float*    dLds   = (float*)(smem + ACC_BYTES + W_BYTES);
    float*    denLds = (float*)(smem + ACC_BYTES + W_BYTES + D_BYTES);

    const int tid = threadIdx.x;

    // ---- phase 0a: kick off async weight-image copy global -> LDS ----
    // 65536 B as 4096 x b128 chunks; 16 chunks per thread, lane-contiguous.
    // Tracked by ASYNCcnt; waited only at the pre-WMMA barrier, so the copy
    // is hidden behind the edge-scatter phase.
    if (wpre != nullptr) {
        #pragma unroll
        for (int i = 0; i < 16; ++i) {
            const int chunk = tid + i * NTHREADS;          // 0..4095
            const unsigned long long gaddr =
                (unsigned long long)(const void*)(wpre + chunk * 8);
            const unsigned lds_off =
                (unsigned)(unsigned long long)(const void*)(wLds + chunk * 8);
            asm volatile("global_load_async_to_lds_b128 %0, %1, off"
                         :: "v"(lds_off), "v"(gaddr)
                         : "memory");
        }
    }

    // ---- phase 0b: zero accumulators, init den ----
    for (int i = tid; i < VPB * RDIM; i += NTHREADS) accLds[i] = 0.0f;
    for (int i = tid; i < 2 * 16 * 16; i += NTHREADS) dLds[i] = 0.0f;
    if (tid < VPB) denLds[tid] = 1.0f;

    // Fallback: stage + convert weights inline if no pre-swizzled image.
    if (wpre == nullptr) {
        for (int idx = tid; idx < RDIM * COUT; idx += NTHREADS) {
            const int co   = idx & (COUT - 1);
            const int r    = idx >> 5;
            const int kk   = r >> 5;
            const int rl   = r & 31;
            const int half = rl >> 4;
            const int slot = rl & 15;
            const int lane = half * 16 + (co & 15);
            const int ntile = co >> 4;
            wLds[((kk * 2 + ntile) * 32 + lane) * 16 + slot] = (_Float16)kern[idx];
        }
    }
    __syncthreads();

    // ---- phase 1: edge scatter into accLds ----
    // One 16-lane group per voxel; lane = input channel.
    const int g  = tid >> 4;
    const int c  = tid & 15;
    const int gv = blockIdx.x * VPB + g;
    if (gv < n_out) {
        const int e0 = rowsplits[gv];
        const int e1 = rowsplits[gv + 1];
        const float px = out_points[gv * 3 + 0];
        const float py = out_points[gv * 3 + 1];
        const float pz = out_points[gv * 3 + 2];
        const float inv = 1.0f / (0.5f * out_extents[gv]);
        float impSum = 0.0f;
        float* acc = accLds + g * RDIM;
        for (int e = e0; e < e1; ++e) {
            const int nb = nidx[e];
            const float r0 = (inp_points[nb * 3 + 0] - px) * inv;
            const float r1 = (inp_points[nb * 3 + 1] - py) * inv;
            const float r2 = (inp_points[nb * 3 + 2] - pz) * inv;
            const float l2   = sqrtf(r0 * r0 + r1 * r1 + r2 * r2);
            const float linf = fmaxf(fmaxf(fabsf(r0), fabsf(r1)), fabsf(r2));
            const float s    = (linf > 0.0f) ? (l2 / fmaxf(linf, 1e-12f)) : 0.0f;
            const float q0 = fminf(fmaxf(r0 * s, -1.0f), 1.0f);
            const float q1 = fminf(fmaxf(r1 * s, -1.0f), 1.0f);
            const float q2 = fminf(fmaxf(r2 * s, -1.0f), 1.0f);
            const float t0 = (q0 + 1.0f) * (0.5f * (KS - 1));
            const float t1 = (q1 + 1.0f) * (0.5f * (KS - 1));
            const float t2 = (q2 + 1.0f) * (0.5f * (KS - 1));
            const int i00 = min(max((int)floorf(t0), 0), KS - 2);
            const int i01 = min(max((int)floorf(t1), 0), KS - 2);
            const int i02 = min(max((int)floorf(t2), 0), KS - 2);
            const float f0 = t0 - (float)i00;
            const float f1 = t1 - (float)i01;
            const float f2 = t2 - (float)i02;
            const float omd = 1.0f - ndist[e];
            const float wp  = fminf(fmaxf(omd * omd * omd, 0.0f), 1.0f);
            const float imp = scale_compat[e] * wp;
            impSum += imp;
            const float psi = imp * feats[nb * CIN + c];
            const float w0a = 1.0f - f0, w1a = 1.0f - f1, w2a = 1.0f - f2;
            #pragma unroll
            for (int cz = 0; cz < 2; ++cz)
                #pragma unroll
                for (int cy = 0; cy < 2; ++cy)
                    #pragma unroll
                    for (int cx = 0; cx < 2; ++cx) {
                        const float wc = (cz ? f0 : w0a) * (cy ? f1 : w1a) * (cx ? f2 : w2a);
                        const int tap = ((i00 + cz) * KS + (i01 + cy)) * KS + (i02 + cx);
                        atomicAdd(&acc[tap * CIN + c], wc * psi);  // ds_add_f32
                    }
        }
        if (c == 0 && impSum != 0.0f) denLds[g] = impSum;
    }

    // Weight image must be resident in LDS before the WMMA phase: each wave
    // drains its own async copies, then the workgroup barrier publishes them.
    if (wpre != nullptr) {
#if __has_builtin(__builtin_amdgcn_s_wait_asynccnt)
        __builtin_amdgcn_s_wait_asynccnt(0);
#else
        asm volatile("s_wait_asynccnt 0x0" ::: "memory");
#endif
    }
    __syncthreads();

    // ---- phase 2: WMMA contraction D[16,32] = acc[16,1024] x W[1024,32] ----
    // 8 waves: wave&1 -> N-tile, wave>>1 -> K quarter (8 of 32 K-steps each).
    {
        const int wave  = tid >> 5;
        const int lane  = tid & 31;
        const int half  = lane >> 4;
        const int m     = lane & 15;     // D row == local voxel
        const int ntile = wave & 1;
        const int kq    = wave >> 1;
        v8f cacc = {};
        for (int s = 0; s < 8; ++s) {
            const int kk = kq * 8 + s;
            // A fragment (16-bit A 16x32): lane half 0 holds K {0..15},
            // half 1 holds K {16..31}, two per dword.
            v16h a = {};
            const float* arow = accLds + m * RDIM + kk * 32;
            #pragma unroll
            for (int j = 0; j < 8; ++j) {
                const int kb = 16 * half + 2 * j;
                a[2 * j]     = (_Float16)arow[kb];
                a[2 * j + 1] = (_Float16)arow[kb + 1];
            }
            // B fragment: one aligned 32B LDS vector load (pre-swizzled)
            const v16h b = *(const v16h*)(wLds + ((kk * 2 + ntile) * 32 + lane) * 16);
            cacc = __builtin_amdgcn_wmma_f32_16x16x32_f16(
                false, a, false, b, (short)0, cacc, false, false);
        }
        // reduce the 4 K-partials per N-tile through LDS.
        // D layout: VGPR j -> row M = j + 8*half, col N = lane%16.
        #pragma unroll
        for (int j = 0; j < 8; ++j)
            atomicAdd(&dLds[ntile * 256 + (j + 8 * half) * 16 + m], cacc[j]);
    }
    __syncthreads();

    // ---- phase 3: normalize + bias + relu + store ----
    for (int idx = tid; idx < VPB * COUT; idx += NTHREADS) {
        const int v  = idx >> 5;
        const int co = idx & 31;
        const int gvo = blockIdx.x * VPB + v;
        if (gvo < n_out) {
            const float val = dLds[(co >> 4) * 256 + v * 16 + (co & 15)];
            const float y   = val / denLds[v] + bias[co];
            out[gvo * COUT + co] = fmaxf(y, 0.0f);
        }
    }
}

extern "C" void kernel_launch(void* const* d_in, const int* in_sizes, int n_in,
                              void* d_out, int out_size, void* d_ws, size_t ws_size,
                              hipStream_t stream) {
    const float* feats        = (const float*)d_in[0];
    const float* inp_points   = (const float*)d_in[1];
    const float* out_points   = (const float*)d_in[2];
    const float* out_extents  = (const float*)d_in[3];
    const float* scale_compat = (const float*)d_in[4];
    const float* ndist        = (const float*)d_in[5];
    const int*   nidx         = (const int*)d_in[6];
    const int*   rowsplits    = (const int*)d_in[7];
    const float* kern         = (const float*)d_in[8];
    const float* bias         = (const float*)d_in[9];
    float* out = (float*)d_out;

    const int n_out  = in_sizes[2] / 3;          // out_points is [N_OUT,3]
    const int blocks = (n_out + VPB - 1) / VPB;

    _Float16* wsw = (ws_size >= (size_t)W_BYTES) ? (_Float16*)d_ws : nullptr;
    if (wsw) {
        hipLaunchKernelGGL(weights_swizzle_kernel, dim3(32), dim3(NTHREADS),
                           0, stream, kern, wsw);
    }

    hipLaunchKernelGGL(cconv_fused_kernel, dim3(blocks), dim3(NTHREADS),
                       LDS_BYTES, stream,
                       feats, inp_points, out_points, out_extents, scale_compat,
                       ndist, nidx, rowsplits, kern, bias, wsw, out, n_out);
}
